// MiniMaxM2MoE_36017595744842
// MI455X (gfx1250) — matrix operand order repaired and checked by reading, not compiled, not run on hardware
//
#include <hip/hip_runtime.h>
#include <math.h>

// Problem constants (from reference)
#define T 4096
#define H 2048
#define E 16
#define I 1024
#define KSEL 4

// GEMM tiling
#define BM 128          // tokens per block tile
#define BN 64           // output columns per block tile
#define KB 64           // K-chunk staged in LDS per iteration
#define LDK (KB + 8)    // padded LDS row stride (ushorts)

typedef __attribute__((ext_vector_type(16))) __bf16          v16bf;
typedef __attribute__((ext_vector_type(8)))  float           v8f;
typedef __attribute__((ext_vector_type(16))) unsigned short  u16x16;

// ---------------------------------------------------------------- helpers

__device__ __forceinline__ unsigned short f2bf(float f) {
    union { float f; unsigned u; } v; v.f = f;
    unsigned r = v.u + 0x7FFFu + ((v.u >> 16) & 1u);   // round-to-nearest-even
    return (unsigned short)(r >> 16);
}

__device__ __forceinline__ void cvt_store8(unsigned short* dst, float4 a, float4 b) {
    union { unsigned short s[8]; uint4 v; } u;
    u.s[0] = f2bf(a.x); u.s[1] = f2bf(a.y); u.s[2] = f2bf(a.z); u.s[3] = f2bf(a.w);
    u.s[4] = f2bf(b.x); u.s[5] = f2bf(b.y); u.s[6] = f2bf(b.z); u.s[7] = f2bf(b.w);
    *(uint4*)dst = u.v;
}

// CDNA5 async global->LDS: copy 64B per lane (4 x b128), ASYNCcnt-tracked.
// IOFFSET is added to BOTH the global and LDS addresses (ISA 08 §4.4), so one
// base pair serves all four transfers. Low 32 bits of a generic pointer to
// __shared__ are the LDS address (ISA §10.2 aperture mapping).
__device__ __forceinline__ void async_copy64B(const unsigned short* gsrc,
                                              unsigned short* ldst) {
    unsigned long long ga = (unsigned long long)gsrc;
    unsigned           la = (unsigned)(unsigned long long)ldst;
    asm volatile(
        "global_load_async_to_lds_b128 %0, %1, off\n\t"
        "global_load_async_to_lds_b128 %0, %1, off offset:16\n\t"
        "global_load_async_to_lds_b128 %0, %1, off offset:32\n\t"
        "global_load_async_to_lds_b128 %0, %1, off offset:48"
        :: "v"(la), "v"(ga) : "memory");
}

__device__ __forceinline__ void wait_async_all() {
    asm volatile("s_wait_asynccnt 0x0" ::: "memory");
}

// A fragment: 16x32 bf16 (lanes 0-15: M=lane, K=0..15;
// lanes 16-31: M=lane-16, K=16..31)
__device__ __forceinline__ v16bf load_a_frag(const unsigned short* base, int ld,
                                             int lane, int kk) {
    int m  = lane & 15;
    int k0 = kk + ((lane >> 4) << 4);
    const unsigned short* p0 = base + m * ld + k0;
    const unsigned short* p1 = p0 + 8;
    u16x16 t;
#pragma unroll
    for (int i = 0; i < 8; ++i) { t[i] = p0[i]; t[8 + i] = p1[i]; }
    return __builtin_bit_cast(v16bf, t);
}

// B fragment: 32x16 bf16 (KxN), lanes 0-15: N=lane, K=0..15; lanes 16-31: K=16..31.
// LDS tile stored [n][k] so each lane reads 16 contiguous bf16.
__device__ __forceinline__ v16bf load_b_frag(const unsigned short* base, int ld,
                                             int lane, int kk) {
    int n  = lane & 15;
    int k0 = kk + ((lane >> 4) << 4);
    const unsigned short* p = base + n * ld + k0;
    u16x16 t;
#pragma unroll
    for (int i = 0; i < 16; ++i) t[i] = p[i];
    return __builtin_bit_cast(v16bf, t);
}

// ---------------------------------------------------------------- kernel 0: init

__global__ void moe_init_kernel(int* __restrict__ counts) {
    if (threadIdx.x < E) counts[threadIdx.x] = 0;
}

// ---------------------------------------------------------------- kernel 1: router

__global__ __launch_bounds__(128) void moe_router_kernel(
    const float* __restrict__ x, const float* __restrict__ gate_w,
    const float* __restrict__ bias, float* __restrict__ out,
    unsigned short* __restrict__ x_bf, int* __restrict__ counts,
    int* __restrict__ tokidx, float* __restrict__ wslot) {

    const int t   = blockIdx.x;
    const int tid = threadIdx.x;
    const int h0  = tid * 16;

    __shared__ float red[E][128];
    __shared__ float sScore[E];
    __shared__ float sChoice[E];

    const float4* xp = (const float4*)(x + (size_t)t * H + h0);
    float4 a0 = xp[0], a1 = xp[1], a2 = xp[2], a3 = xp[3];
    float xv[16] = { a0.x,a0.y,a0.z,a0.w, a1.x,a1.y,a1.z,a1.w,
                     a2.x,a2.y,a2.z,a2.w, a3.x,a3.y,a3.z,a3.w };
    union { unsigned short s[16]; uint4 v[2]; } pk;
#pragma unroll
    for (int i = 0; i < 16; ++i) pk.s[i] = f2bf(xv[i]);
    uint4* xb = (uint4*)(x_bf + (size_t)t * H + h0);
    xb[0] = pk.v[0]; xb[1] = pk.v[1];
#pragma unroll
    for (int i = 0; i < 16; ++i) out[(size_t)t * H + h0 + i] = 0.0f;

#pragma unroll
    for (int e = 0; e < E; ++e) {
        const float4* gp = (const float4*)(gate_w + (size_t)e * H + h0);
        float4 g0 = gp[0], g1 = gp[1], g2 = gp[2], g3 = gp[3];
        float acc = a0.x*g0.x + a0.y*g0.y + a0.z*g0.z + a0.w*g0.w
                  + a1.x*g1.x + a1.y*g1.y + a1.z*g1.z + a1.w*g1.w
                  + a2.x*g2.x + a2.y*g2.y + a2.z*g2.z + a2.w*g2.w
                  + a3.x*g3.x + a3.y*g3.y + a3.z*g3.z + a3.w*g3.w;
        red[e][tid] = acc;
    }
    __syncthreads();

    if (tid < E) {
        float s = 0.0f;
        for (int i = 0; i < 128; ++i) s += red[tid][i];
        float sc = 1.0f / (1.0f + __expf(-s));
        sScore[tid]  = sc;
        sChoice[tid] = sc + bias[tid];
    }
    __syncthreads();

    if (tid == 0) {
        float ch[E], sc[E];
#pragma unroll
        for (int e = 0; e < E; ++e) { ch[e] = sChoice[e]; sc[e] = sScore[e]; }
        int   sel[KSEL];
        float wsum = 0.0f;
#pragma unroll
        for (int k = 0; k < KSEL; ++k) {
            int   best = 0;
            float bv   = -3.0e38f;
            for (int e = 0; e < E; ++e)
                if (ch[e] > bv) { bv = ch[e]; best = e; }
            sel[k]   = best;
            ch[best] = -3.0e38f;
            wsum    += sc[best];
        }
        float inv = 1.0f / wsum;
#pragma unroll
        for (int k = 0; k < KSEL; ++k) {
            int e    = sel[k];
            int slot = atomicAdd(&counts[e], 1);
            tokidx[(size_t)e * T + slot] = t;
            wslot [(size_t)e * T + slot] = sc[e] * inv;
        }
    }
}

// ---------------------------------------------------------------- kernel 2: scan

__global__ void moe_scan_kernel(const int* __restrict__ counts,
                                int* __restrict__ offsets) {
    if (threadIdx.x == 0) {
        int acc = 0;
        for (int e = 0; e < E; ++e) { offsets[e] = acc; acc += counts[e]; }
    }
}

// ---------------------------------------------------------------- kernel 3: gate/up
// grid = (I/BN, T/BM, E). Double-buffered LDS; A staged with async global->LDS
// (bf16, no conversion), B staged through VGPRs with fp32->bf16 conversion.

__global__ __launch_bounds__(256) void moe_gateup_kernel(
    const unsigned short* __restrict__ x_bf,
    const float* __restrict__ w_gate, const float* __restrict__ w_up,
    const int* __restrict__ counts, const int* __restrict__ offsets,
    const int* __restrict__ tokidx, const float* __restrict__ wslot,
    unsigned short* __restrict__ h2) {

    const int e   = blockIdx.z;
    const int mt  = blockIdx.y;
    const int nt  = blockIdx.x;
    const int cnt = counts[e];
    if (mt * BM >= cnt) return;
    const int rem     = cnt - mt * BM;
    const int rowBase = offsets[e] + mt * BM;

    const int tid  = threadIdx.x;
    const int lane = tid & 31;
    const int wv   = tid >> 5;            // 8 waves: 4 (M) x 2 (N)
    const int wm   = (wv & 3) * 32;
    const int wn   = (wv >> 2) * 32;

    __shared__ unsigned short As[2][BM][LDK];
    __shared__ unsigned short Bg[2][BN][LDK];
    __shared__ unsigned short Bu[2][BN][LDK];
    __shared__ int   sTok[BM];
    __shared__ float sW[BM];

    if (tid < BM) {
        int idx = mt * BM + tid;
        sTok[tid] = (idx < cnt) ? tokidx[(size_t)e * T + idx] : tokidx[(size_t)e * T];
        sW[tid]   = (idx < cnt) ? wslot [(size_t)e * T + idx] : 0.0f;
    }
    __syncthreads();

    const int arow = tid >> 1, ac0 = (tid & 1) * 32;                       // 64B/thread
    const int brow = tid >> 2, bc0 = (tid & 2 ? 32 : 0) + (tid & 1) * 16;  // 16 floats

    const unsigned short* aSrc = x_bf + (size_t)sTok[arow] * H + ac0;
    const size_t bBase = ((size_t)e * I + nt * BN + brow) * H + bc0;

    v8f accG[2][2] = {{{}, {}}, {{}, {}}};
    v8f accU[2][2] = {{{}, {}}, {{}, {}}};
    float4 pg[4], pu[4];

    // prologue: stage k=0 into buffer 0
    async_copy64B(aSrc, &As[0][arow][ac0]);
    {
        const float4* g = (const float4*)(w_gate + bBase);
        const float4* u = (const float4*)(w_up + bBase);
#pragma unroll
        for (int i = 0; i < 4; ++i) { pg[i] = g[i]; pu[i] = u[i]; }
        cvt_store8(&Bg[0][brow][bc0],     pg[0], pg[1]);
        cvt_store8(&Bg[0][brow][bc0 + 8], pg[2], pg[3]);
        cvt_store8(&Bu[0][brow][bc0],     pu[0], pu[1]);
        cvt_store8(&Bu[0][brow][bc0 + 8], pu[2], pu[3]);
    }
    wait_async_all();
    __syncthreads();

    for (int k = 0; k < H; k += KB) {
        const int  cur  = (k / KB) & 1;
        const int  nxt  = cur ^ 1;
        const bool more = (k + KB) < H;

        if (more) {
            async_copy64B(aSrc + (k + KB), &As[nxt][arow][ac0]);
            const float4* g = (const float4*)(w_gate + bBase + (k + KB));
            const float4* u = (const float4*)(w_up + bBase + (k + KB));
#pragma unroll
            for (int i = 0; i < 4; ++i) { pg[i] = g[i]; pu[i] = u[i]; }
        }

#pragma unroll
        for (int kk = 0; kk < KB; kk += 32) {
            v16bf aF[2], gF[2], uF[2];
#pragma unroll
            for (int i = 0; i < 2; ++i) {
                aF[i] = load_a_frag(&As[cur][wm + 16 * i][0], LDK, lane, kk);
                gF[i] = load_b_frag(&Bg[cur][wn + 16 * i][0], LDK, lane, kk);
                uF[i] = load_b_frag(&Bu[cur][wn + 16 * i][0], LDK, lane, kk);
            }
#pragma unroll
            for (int i = 0; i < 2; ++i)
#pragma unroll
                for (int j = 0; j < 2; ++j) {
                    accG[i][j] = __builtin_amdgcn_wmma_f32_16x16x32_bf16(
                        false, aF[i], false, gF[j], (short)0, accG[i][j], false, false);
                    accU[i][j] = __builtin_amdgcn_wmma_f32_16x16x32_bf16(
                        false, aF[i], false, uF[j], (short)0, accU[i][j], false, false);
                }
        }

        if (more) {
            cvt_store8(&Bg[nxt][brow][bc0],     pg[0], pg[1]);
            cvt_store8(&Bg[nxt][brow][bc0 + 8], pg[2], pg[3]);
            cvt_store8(&Bu[nxt][brow][bc0],     pu[0], pu[1]);
            cvt_store8(&Bu[nxt][brow][bc0 + 8], pu[2], pu[3]);
        }
        wait_async_all();
        __syncthreads();
    }

    // epilogue: h2 = silu(g) * u * cw, store bf16
    const int colL    = lane & 15;
    const int rowHalf = (lane >> 4) << 3;
#pragma unroll
    for (int i = 0; i < 2; ++i)
#pragma unroll
        for (int j = 0; j < 2; ++j)
#pragma unroll
            for (int r = 0; r < 8; ++r) {
                int m = wm + 16 * i + rowHalf + r;
                int n = wn + 16 * j + colL;
                if (m < rem) {
                    float g = accG[i][j][r];
                    float u = accU[i][j][r];
                    float hval = (g / (1.0f + __expf(-g))) * u * sW[m];
                    h2[(size_t)(rowBase + m) * I + nt * BN + n] = f2bf(hval);
                }
            }
}

// ---------------------------------------------------------------- kernel 4: down
// grid = (H/BN, T/BM, E). out[tok, n] += h2_row · Wd[e][n, :] via atomic f32 adds.

__global__ __launch_bounds__(256) void moe_down_kernel(
    const unsigned short* __restrict__ h2, const float* __restrict__ w_down,
    const int* __restrict__ counts, const int* __restrict__ offsets,
    const int* __restrict__ tokidx, float* __restrict__ out) {

    const int e   = blockIdx.z;
    const int mt  = blockIdx.y;
    const int nt  = blockIdx.x;
    const int cnt = counts[e];
    if (mt * BM >= cnt) return;
    const int rem     = cnt - mt * BM;
    const int rowBase = offsets[e] + mt * BM;

    const int tid  = threadIdx.x;
    const int lane = tid & 31;
    const int wv   = tid >> 5;
    const int wm   = (wv & 3) * 32;
    const int wn   = (wv >> 2) * 32;

    __shared__ unsigned short As[2][BM][LDK];
    __shared__ unsigned short Bd[2][BN][LDK];
    __shared__ int sTok[BM];

    if (tid < BM) {
        int idx = mt * BM + tid;
        sTok[tid] = (idx < cnt) ? tokidx[(size_t)e * T + idx] : 0;
    }
    __syncthreads();

    const int arow = tid >> 1, ac0 = (tid & 1) * 32;
    const int brow = tid >> 2, bc0 = (tid & 2 ? 32 : 0) + (tid & 1) * 16;

    const unsigned short* aSrc = h2 + (size_t)(rowBase + arow) * I + ac0;
    const size_t bBase = ((size_t)e * H + nt * BN + brow) * I + bc0;

    v8f acc[2][2] = {{{}, {}}, {{}, {}}};
    float4 pd[4];

    // prologue
    async_copy64B(aSrc, &As[0][arow][ac0]);
    {
        const float4* d = (const float4*)(w_down + bBase);
#pragma unroll
        for (int i = 0; i < 4; ++i) pd[i] = d[i];
        cvt_store8(&Bd[0][brow][bc0],     pd[0], pd[1]);
        cvt_store8(&Bd[0][brow][bc0 + 8], pd[2], pd[3]);
    }
    wait_async_all();
    __syncthreads();

    for (int k = 0; k < I; k += KB) {
        const int  cur  = (k / KB) & 1;
        const int  nxt  = cur ^ 1;
        const bool more = (k + KB) < I;

        if (more) {
            async_copy64B(aSrc + (k + KB), &As[nxt][arow][ac0]);
            const float4* d = (const float4*)(w_down + bBase + (k + KB));
#pragma unroll
            for (int i = 0; i < 4; ++i) pd[i] = d[i];
        }

#pragma unroll
        for (int kk = 0; kk < KB; kk += 32) {
            v16bf aF[2], bF[2];
#pragma unroll
            for (int i = 0; i < 2; ++i) {
                aF[i] = load_a_frag(&As[cur][wm + 16 * i][0], LDK, lane, kk);
                bF[i] = load_b_frag(&Bd[cur][wn + 16 * i][0], LDK, lane, kk);
            }
#pragma unroll
            for (int i = 0; i < 2; ++i)
#pragma unroll
                for (int j = 0; j < 2; ++j)
                    acc[i][j] = __builtin_amdgcn_wmma_f32_16x16x32_bf16(
                        false, aF[i], false, bF[j], (short)0, acc[i][j], false, false);
        }

        if (more) {
            cvt_store8(&Bd[nxt][brow][bc0],     pd[0], pd[1]);
            cvt_store8(&Bd[nxt][brow][bc0 + 8], pd[2], pd[3]);
        }
        wait_async_all();
        __syncthreads();
    }

    const int colL    = lane & 15;
    const int rowHalf = (lane >> 4) << 3;
#pragma unroll
    for (int i = 0; i < 2; ++i)
#pragma unroll
        for (int j = 0; j < 2; ++j)
#pragma unroll
            for (int r = 0; r < 8; ++r) {
                int m = wm + 16 * i + rowHalf + r;
                int n = wn + 16 * j + colL;
                if (m < rem)
                    atomicAdd(&out[(size_t)sTok[m] * H + nt * BN + n], acc[i][j][r]);
            }
}

// ---------------------------------------------------------------- launch

extern "C" void kernel_launch(void* const* d_in, const int* in_sizes, int n_in,
                              void* d_out, int out_size, void* d_ws, size_t ws_size,
                              hipStream_t stream) {
    const float* x      = (const float*)d_in[0];
    const float* gate_w = (const float*)d_in[1];
    const float* bias   = (const float*)d_in[2];
    const float* w_gate = (const float*)d_in[3];
    const float* w_up   = (const float*)d_in[4];
    const float* w_down = (const float*)d_in[5];
    float* out = (float*)d_out;

    // workspace layout (bytes)
    char* ws = (char*)d_ws;
    int*   counts  = (int*)ws;                                   // 64 B
    int*   offsets = (int*)(ws + 256);                           // 64 B
    int*   tokidx  = (int*)(ws + 512);                           // E*T*4   = 256 KB
    float* wslot   = (float*)(ws + 512 + (size_t)E * T * 4);     // E*T*4   = 256 KB
    unsigned short* x_bf = (unsigned short*)(ws + 512 + 2 * (size_t)E * T * 4); // T*H*2 = 16 MB
    unsigned short* h2   = x_bf + (size_t)T * H;                 // (T*K+256)*I*2 ≈ 34 MB

    moe_init_kernel<<<1, 32, 0, stream>>>(counts);
    moe_router_kernel<<<T, 128, 0, stream>>>(x, gate_w, bias, out, x_bf,
                                             counts, tokidx, wslot);
    moe_scan_kernel<<<1, 32, 0, stream>>>(counts, offsets);

    dim3 g1(I / BN, T / BM, E);   // (16, 32, 16)
    moe_gateup_kernel<<<g1, 256, 0, stream>>>(x_bf, w_gate, w_up, counts, offsets,
                                              tokidx, wslot, h2);

    dim3 g2(H / BN, T / BM, E);   // (32, 32, 16)
    moe_down_kernel<<<g2, 256, 0, stream>>>(h2, w_down, counts, offsets, tokidx, out);
}
